// QueryConditionedMoE_12824772346168
// MI455X (gfx1250) — compile-verified, matrix operand order checked by reading, not run on hardware
//
#include <hip/hip_runtime.h>
#include <hip/hip_bf16.h>
#include <math.h>

#define HID 1024
#define QRY 1024
#define TOK 16384      // 64*256 tokens
#define MT  32         // tokens per workgroup tile
#define NVE 12         // 4 shared-chunks + 8 routed experts

typedef _Float16 vh16 __attribute__((ext_vector_type(16)));
typedef _Float16 vh8  __attribute__((ext_vector_type(8)));
typedef float    vf8  __attribute__((ext_vector_type(8)));
typedef unsigned int vu4 __attribute__((ext_vector_type(4)));
typedef int      vi4 __attribute__((ext_vector_type(4)));
typedef int      vi8 __attribute__((ext_vector_type(8)));

#if defined(__has_builtin)
#if __has_builtin(__builtin_amdgcn_tensor_load_to_lds)
#define USE_TDM 1
#endif
#endif

__device__ __forceinline__ vf8 zero8() {
  vf8 z;
#pragma unroll
  for (int i = 0; i < 8; ++i) z[i] = 0.0f;
  return z;
}

// Load one 16-element f16 fragment following the CDNA5 16-bit A/B layout:
// lane<16 holds K {k..k+7, k+16..k+23}; lane>=16 holds K {k+8..k+15, k+24..k+31}.
// Caller passes p = rowBase + k + ((lane>>4)<<3); we fetch 8 halfs at p and 8 at p+16.
__device__ __forceinline__ vh16 frag_pair(const _Float16* p) {
  vh8 lo = *(const vh8*)(p);
  vh8 hi = *(const vh8*)(p + 16);
  return __builtin_shufflevector(lo, hi, 0, 1, 2, 3, 4, 5, 6, 7,
                                 8, 9, 10, 11, 12, 13, 14, 15);
}

__device__ __forceinline__ vf8 wmma_f16(vh16 a, vh16 b, vf8 c) {
  // emits v_wmma_f32_16x16x32_f16
  return __builtin_amdgcn_wmma_f32_16x16x32_f16(false, a, false, b,
                                                (short)0, c, false, false);
}

// Branch-free GELU (tanh form): v_exp_f32 + v_rcp_f32 + FMAs only.
// The transcendental pipe co-executes with WMMA, so this stays off the
// critical path; no exec-mask divergence like device-libm erff.
__device__ __forceinline__ float gelu_fast(float x) {
  const float u = 0.7978845608028654f * x * (1.0f + 0.044715f * x * x);
  const float t = 1.0f - 2.0f * __builtin_amdgcn_rcpf(1.0f + __expf(2.0f * u));
  return 0.5f * x * (1.0f + t);
}

// ---------------------------------------------------------------------------
// K1: FiLM MLP (4 rows, tiny) + qdot precompute + zero loss accumulators
// ---------------------------------------------------------------------------
__global__ __launch_bounds__(256)
void film_kernel(const float* __restrict__ query,
                 const float* __restrict__ w1, const float* __restrict__ b1,
                 const float* __restrict__ w2, const float* __restrict__ b2,
                 const float* __restrict__ gate_w,
                 float* __restrict__ gamma, float* __restrict__ beta,
                 float* __restrict__ qdot, float* __restrict__ lossAcc) {
  __shared__ float sh[4][2048];
  const int tid = threadIdx.x;
  if (tid < 17) lossAcc[tid] = 0.0f;  // lsum[8] | psum[8] | zsum[1]
  for (int idx = tid; idx < 4 * 2048; idx += 256) {
    const int bq = idx >> 11, j = idx & 2047;
    const float* q = query + bq * QRY;
    const float* w = w1 + (size_t)j * QRY;
    float a = b1[j];
    for (int k = 0; k < QRY; ++k) a += q[k] * w[k];
    sh[bq][j] = a * __builtin_amdgcn_rcpf(1.0f + __expf(-a));  // SiLU
  }
  __syncthreads();
  for (int idx = tid; idx < 4 * 2048; idx += 256) {
    const int bq = idx >> 11, j = idx & 2047;
    const float* w = w2 + (size_t)j * 2048;
    float a = b2[j];
    for (int k = 0; k < 2048; ++k) a += sh[bq][k] * w[k];
    if (j < HID) gamma[bq * HID + j] = a;
    else         beta[bq * HID + (j - HID)] = a;
  }
  if (tid < 32) {  // qdot[b][e] = q[b] . gate_w[e, HID:2*HID]
    const int bq = tid >> 3, e = tid & 7;
    const float* q = query + bq * QRY;
    const float* w = gate_w + e * (HID + QRY) + HID;
    float a = 0.0f;
    for (int k = 0; k < QRY; ++k) a += q[k] * w[k];
    qdot[tid] = a;
  }
}

// ---------------------------------------------------------------------------
// K2: f32 -> f16 weight conversion (weights then live in L2 as f16)
// ---------------------------------------------------------------------------
__global__ __launch_bounds__(256)
void cvt_f16_kernel(const float* __restrict__ s, _Float16* __restrict__ d, int n) {
  for (int i = blockIdx.x * 256 + threadIdx.x; i < n; i += gridDim.x * 256)
    d[i] = (_Float16)s[i];
}

// ---------------------------------------------------------------------------
// K3: FiLM conditioning + router (softmax, top-2, comb weights, loss partials)
// one wave per token
// ---------------------------------------------------------------------------
__global__ __launch_bounds__(256)
void cond_gate_kernel(const float* __restrict__ vis,
                      const float* __restrict__ gate_w,
                      const float* __restrict__ gamma,
                      const float* __restrict__ beta,
                      const float* __restrict__ qdot,
                      _Float16* __restrict__ Xc,
                      float* __restrict__ comb,
                      float* __restrict__ lsum,
                      float* __restrict__ psum,
                      float* __restrict__ zsum) {
  const int token = blockIdx.x * 8 + (threadIdx.x >> 5);
  const int lane = threadIdx.x & 31;
  const int qb = token >> 12;  // 4096 tokens per query row
  const float* g = gamma + qb * HID;
  const float* b = beta + qb * HID;
  const float* v = vis + (size_t)token * HID;
  _Float16* xrow = Xc + (size_t)token * HID;

  float lg[8];
#pragma unroll
  for (int e = 0; e < 8; ++e) lg[e] = 0.0f;
#pragma unroll 4
  for (int i = 0; i < 32; ++i) {
    const int idx = lane + 32 * i;
    const float c = g[idx] * v[idx] + b[idx];
    xrow[idx] = (_Float16)c;
#pragma unroll
    for (int e = 0; e < 8; ++e) lg[e] += c * gate_w[e * (HID + QRY) + idx];
  }
#pragma unroll
  for (int e = 0; e < 8; ++e)
    for (int off = 16; off > 0; off >>= 1) lg[e] += __shfl_xor(lg[e], off, 32);
  if (lane == 0) {
    float mx = -1e30f;
#pragma unroll
    for (int e = 0; e < 8; ++e) { lg[e] += qdot[qb * 8 + e]; mx = fmaxf(mx, lg[e]); }
    float s = 0.0f, p[8];
#pragma unroll
    for (int e = 0; e < 8; ++e) { p[e] = __expf(lg[e] - mx); s += p[e]; }
    const float inv = __builtin_amdgcn_rcpf(s);
#pragma unroll
    for (int e = 0; e < 8; ++e) p[e] *= inv;
    int i1 = 0;
#pragma unroll
    for (int e = 1; e < 8; ++e) if (p[e] > p[i1]) i1 = e;
    int i2 = (i1 == 0) ? 1 : 0;
#pragma unroll
    for (int e = 0; e < 8; ++e) if (e != i1 && p[e] > p[i2]) i2 = e;
#pragma unroll
    for (int e = 0; e < 8; ++e)
      comb[(size_t)token * 8 + e] = (e == i1) ? p[i1] : ((e == i2) ? p[i2] : 0.0f);
    const float logZ = mx + __logf(s);
    atomicAdd(&lsum[i1], 0.5f);
    atomicAdd(&lsum[i2], 0.5f);
#pragma unroll
    for (int e = 0; e < 8; ++e) atomicAdd(&psum[e], p[e]);
    atomicAdd(zsum, logZ * logZ);
  }
}

// ---------------------------------------------------------------------------
// K4: main WMMA kernel. One workgroup = 32 tokens x full 1024 output dim.
// Loops 12 virtual experts (4 shared chunks w=1, 8 routed w=comb[t,e]):
//   H = rowscale(gelu(X @ Wup^T))  -> LDS (f16)
//   acc += H @ Wdown^T             (f32 WMMA accumulators, persistent)
// Epilogue: residual + LayerNorm via LDS row reductions.
// ---------------------------------------------------------------------------
__global__ __launch_bounds__(256, 1)
void moe_main_kernel(const float* __restrict__ vis,
                     const _Float16* __restrict__ Xc,
                     const float* __restrict__ comb,
                     const _Float16* __restrict__ swupH,
                     const _Float16* __restrict__ swdnH,
                     const _Float16* __restrict__ ewupH,
                     const _Float16* __restrict__ ewdnH,
                     const float* __restrict__ lng,
                     const float* __restrict__ lnb,
                     float* __restrict__ out) {
  extern __shared__ char smem[];
  _Float16* sX = (_Float16*)smem;                    // [32][1024] f16 (64KB)
  _Float16* sH = (_Float16*)(smem + 64 * 1024);      // [32][1024] f16 (64KB)
  float* sScale = (float*)(smem + 128 * 1024);       // [NVE][32] row scales (1.5KB)
  float* sSum = (float*)(smem + 128 * 1024 + 1536);  // [32] + [32] (contig, 256B)

  const int tid = threadIdx.x;
  const int wave = tid >> 5;
  const int lane = tid & 31;
  const int row0 = blockIdx.x * MT;
  const int mHalf = wave >> 2;   // rows [16*mHalf, +16)
  const int nChunk = wave & 3;   // cols [256*nChunk, +256)
  const int lmod = lane & 15;
  const int lhi8 = (lane >> 4) << 3;

  // ---- stage X tile (32x1024 f16) into LDS with the Tensor Data Mover ----
#ifdef USE_TDM
  if (wave == 0) {
    const unsigned long long ga =
        (unsigned long long)(const void*)(Xc + (size_t)row0 * HID);
    vu4 g0; vi8 g1; vi4 g2; vi4 g3;
    g0[0] = 1u;                                           // count=1 (user D#)
    g0[1] = 0u;                                           // lds_addr: dyn-LDS base (sX)
    g0[2] = (unsigned)(ga & 0xffffffffull);               // global_addr lo
    g0[3] = (unsigned)((ga >> 32) & 0x01fffffful) | 0x80000000u;  // addr hi | type=2
    g1[0] = 0x00010000;                                   // data_size = 2 bytes
    g1[1] = (int)(((unsigned)HID & 0xffffu) << 16);       // tensor_dim0 = 1024
    g1[2] = (int)(((unsigned)MT & 0xffffu) << 16);        // tensor_dim1 = 32
    g1[3] = (int)(((unsigned)HID & 0xffffu) << 16);       // tile_dim0 = 1024
    g1[4] = MT;                                           // tile_dim1 = 32, tile_dim2 = 0
    g1[5] = HID;                                          // tensor_dim0_stride = 1024
    g1[6] = 0; g1[7] = 0;
#pragma unroll
    for (int i = 0; i < 4; ++i) { g2[i] = 0; g3[i] = 0; }
#if __clang_major__ >= 23
    vi8 g4;
#pragma unroll
    for (int i = 0; i < 8; ++i) g4[i] = 0;
    __builtin_amdgcn_tensor_load_to_lds(g0, g1, g2, g3, g4, 0);
#else
    __builtin_amdgcn_tensor_load_to_lds(g0, g1, g2, g3, 0);
#endif
    __builtin_amdgcn_s_wait_tensorcnt(0);
  }
#else
  {
    const vh8* src = (const vh8*)(Xc + (size_t)row0 * HID);
    vh8* dst = (vh8*)sX;
    for (int i = tid; i < MT * HID / 8; i += 256) dst[i] = src[i];
  }
#endif
  // Per-(virtual expert, row) scale table: shared chunks get 1.0, routed get
  // comb[token, e].  Flat unconditional loads in the epilogue -> no divergence.
  for (int i = tid; i < NVE * MT; i += 256) {
    const int ve = i >> 5, m = i & 31;
    sScale[i] = (ve < 4) ? 1.0f : comb[(size_t)(row0 + m) * 8 + (ve - 4)];
  }
  if (tid < 64) sSum[tid] = 0.0f;  // zero sSum[32] + sSum2[32]
  __syncthreads();

  vf8 acc[16];
#pragma unroll
  for (int t = 0; t < 16; ++t) acc[t] = zero8();

  for (int ve = 0; ve < NVE; ++ve) {
    const _Float16* up;
    const _Float16* dn;
    int upStride, dnStride;
    if (ve < 4) {  // shared-expert chunk: inter rows [ve*1024, +1024)
      up = swupH + (size_t)ve * 1024 * HID; upStride = HID;
      dn = swdnH + (size_t)ve * 1024;       dnStride = 4096;
    } else {       // routed expert e
      const int e = ve - 4;
      up = ewupH + (size_t)e * HID * HID;   upStride = HID;
      dn = ewdnH + (size_t)e * HID * HID;   dnStride = HID;
    }
    __builtin_prefetch(up, 0, 1);  // global_prefetch_b8
    __builtin_prefetch(dn, 0, 1);

    // ---- up-projection + GELU + routing scale -> sH ----
#pragma unroll
    for (int t = 0; t < 16; ++t) {
      const int nBase = nChunk * 256 + t * 16;  // inter-dim column
      vf8 c = zero8();
#pragma unroll 2
      for (int k = 0; k < HID; k += 32) {
        vh16 a = frag_pair(sX + (16 * mHalf + lmod) * HID + k + lhi8);
        vh16 b = frag_pair(up + (size_t)(nBase + lmod) * upStride + k + lhi8);
        c = wmma_f16(a, b, c);
      }
#pragma unroll
      for (int r = 0; r < 8; ++r) {
        const int mm = 16 * mHalf + (lane >> 4) * 8 + r;  // C/D layout row
        const float sc = sScale[ve * MT + mm];
        sH[mm * HID + nBase + lmod] = (_Float16)(gelu_fast(c[r]) * sc);
      }
    }
    __syncthreads();

    // ---- down-projection, accumulate into persistent f32 acc ----
#pragma unroll
    for (int t = 0; t < 16; ++t) {
      const int nBase = nChunk * 256 + t * 16;  // output column
      vf8 c = acc[t];
#pragma unroll 2
      for (int k = 0; k < HID; k += 32) {
        vh16 a = frag_pair(sH + (16 * mHalf + lmod) * HID + k + lhi8);
        vh16 b = frag_pair(dn + (size_t)(nBase + lmod) * dnStride + k + lhi8);
        c = wmma_f16(a, b, c);
      }
      acc[t] = c;
    }
    __syncthreads();  // WAR on sH before next expert
  }

  // ---- residual + LayerNorm epilogue ----
  float ps[8], ps2[8];
#pragma unroll
  for (int r = 0; r < 8; ++r) { ps[r] = 0.0f; ps2[r] = 0.0f; }
#pragma unroll
  for (int t = 0; t < 16; ++t) {
    const int n = nChunk * 256 + t * 16 + lmod;
#pragma unroll
    for (int r = 0; r < 8; ++r) {
      const int mm = 16 * mHalf + (lane >> 4) * 8 + r;
      const float x = acc[t][r] + vis[(size_t)(row0 + mm) * HID + n];
      acc[t][r] = x;
      ps[r] += x;
      ps2[r] += x * x;
    }
  }
#pragma unroll
  for (int r = 0; r < 8; ++r) {
    const int mm = 16 * mHalf + (lane >> 4) * 8 + r;
    atomicAdd(&sSum[mm], ps[r]);        // ds_add_f32
    atomicAdd(&sSum[32 + mm], ps2[r]);
  }
  __syncthreads();
#pragma unroll
  for (int t = 0; t < 16; ++t) {
    const int n = nChunk * 256 + t * 16 + lmod;
    const float gn = lng[n], bn = lnb[n];
#pragma unroll
    for (int r = 0; r < 8; ++r) {
      const int mm = 16 * mHalf + (lane >> 4) * 8 + r;
      const float mu = sSum[mm] * (1.0f / HID);
      const float var = sSum[32 + mm] * (1.0f / HID) - mu * mu;
      out[(size_t)(row0 + mm) * HID + n] =
          (acc[t][r] - mu) * rsqrtf(var + 1e-5f) * gn + bn;
    }
  }
}

// ---------------------------------------------------------------------------
// K5: finalize the two scalar losses
// ---------------------------------------------------------------------------
__global__ void finalize_kernel(const float* __restrict__ accv, float* __restrict__ out) {
  if (threadIdx.x == 0 && blockIdx.x == 0) {
    const float inv = 1.0f / 16384.0f;
    float lb = 0.0f;
    for (int e = 0; e < 8; ++e) lb += (accv[e] * inv) * (accv[8 + e] * inv);
    out[(size_t)TOK * HID] = 8.0f * lb;
    out[(size_t)TOK * HID + 1] = accv[16] * inv;
  }
}

// ---------------------------------------------------------------------------
// workspace layout (bytes)
// ---------------------------------------------------------------------------
#define WS_XC    ((size_t)0)            // f16 [16384*1024]  -> 33554432 B
#define WS_SWUP  ((size_t)33554432)     // f16 [4096*1024]   ->  8388608 B
#define WS_SWDN  ((size_t)41943040)     // f16 [1024*4096]   ->  8388608 B
#define WS_EWUP  ((size_t)50331648)     // f16 [8*1024*1024] -> 16777216 B
#define WS_EWDN  ((size_t)67108864)     // f16 [8*1024*1024] -> 16777216 B
#define WS_COMB  ((size_t)83886080)     // f32 [16384*8]     ->   524288 B
#define WS_GAMMA ((size_t)84410368)     // f32 [4*1024]
#define WS_BETA  ((size_t)84426752)     // f32 [4*1024]
#define WS_QDOT  ((size_t)84443136)     // f32 [32]
#define WS_LOSS  ((size_t)84443264)     // f32 [17]: lsum8 | psum8 | zsum1

extern "C" void kernel_launch(void* const* d_in, const int* in_sizes, int n_in,
                              void* d_out, int out_size, void* d_ws, size_t ws_size,
                              hipStream_t stream) {
  (void)in_sizes; (void)n_in; (void)out_size; (void)ws_size;
  const float* vis   = (const float*)d_in[0];
  const float* query = (const float*)d_in[1];
  const float* fw1   = (const float*)d_in[2];
  const float* fb1   = (const float*)d_in[3];
  const float* fw2   = (const float*)d_in[4];
  const float* fb2   = (const float*)d_in[5];
  const float* gw    = (const float*)d_in[6];
  const float* swup  = (const float*)d_in[7];
  const float* swdn  = (const float*)d_in[8];
  const float* ewup  = (const float*)d_in[9];
  const float* ewdn  = (const float*)d_in[10];
  const float* lng   = (const float*)d_in[11];
  const float* lnb   = (const float*)d_in[12];
  float* out = (float*)d_out;
  char* ws = (char*)d_ws;

  _Float16* Xc    = (_Float16*)(ws + WS_XC);
  _Float16* swupH = (_Float16*)(ws + WS_SWUP);
  _Float16* swdnH = (_Float16*)(ws + WS_SWDN);
  _Float16* ewupH = (_Float16*)(ws + WS_EWUP);
  _Float16* ewdnH = (_Float16*)(ws + WS_EWDN);
  float* combP  = (float*)(ws + WS_COMB);
  float* gammaP = (float*)(ws + WS_GAMMA);
  float* betaP  = (float*)(ws + WS_BETA);
  float* qdotP  = (float*)(ws + WS_QDOT);
  float* lossP  = (float*)(ws + WS_LOSS);  // lsum(8) | psum(8) | zsum(1)

  film_kernel<<<1, 256, 0, stream>>>(query, fw1, fb1, fw2, fb2, gw,
                                     gammaP, betaP, qdotP, lossP);
  cvt_f16_kernel<<<2048, 256, 0, stream>>>(swup, swupH, 4096 * 1024);
  cvt_f16_kernel<<<2048, 256, 0, stream>>>(swdn, swdnH, 1024 * 4096);
  cvt_f16_kernel<<<4096, 256, 0, stream>>>(ewup, ewupH, 8 * 1024 * 1024);
  cvt_f16_kernel<<<4096, 256, 0, stream>>>(ewdn, ewdnH, 8 * 1024 * 1024);
  cond_gate_kernel<<<TOK / 8, 256, 0, stream>>>(vis, gw, gammaP, betaP, qdotP,
                                                Xc, combP, lossP, lossP + 8,
                                                lossP + 16);
  moe_main_kernel<<<TOK / MT, 256, 132864, stream>>>(
      vis, Xc, combP, swupH, swdnH, ewupH, ewdnH, lng, lnb, out);
  finalize_kernel<<<1, 32, 0, stream>>>(lossP, out);
}
